// ResCom_71073118814983
// MI455X (gfx1250) — compile-verified, matrix operand order checked by reading, not run on hardware
//
#include <hip/hip_runtime.h>

// ---------------- problem constants (from reference) ----------------
#define N_SAMPLES 2048
#define DIM_IN    1024
#define DIM_FEAT  2048
#define DIM_EMB   128
#define NCLS      1000
#define QSZ       8
#define QC        8000      // Q*C
#define KPOS      4
#define KNEG      1024
#define NEG_LEN   7992      // Q*(C-1)
#define LD_CON    3075      // 2047 offdiag + 4 pos + 1024 neg

typedef __attribute__((ext_vector_type(16))) __bf16 v16bf;
typedef __attribute__((ext_vector_type(8)))  float  v8f;

// f32 -> bf16 bits, round-to-nearest-even
__device__ __forceinline__ unsigned short bfbits(float f) {
  union { float f; unsigned u; } v; v.f = f;
  return (unsigned short)((v.u + 0x7FFFu + ((v.u >> 16) & 1u)) >> 16);
}
__device__ __forceinline__ __bf16 f2bf(float f) {
  unsigned short us = bfbits(f);
  return __builtin_bit_cast(__bf16, us);
}

// ---------------- WMMA fragment load (CDNA5 ISA 7.12.2) ----------------
// Operand stored row-major [row][K] in LDS (A rows = M, B rows = N since B is
// pre-transposed).  Lane layout: lanes 0-15 row=lane, K-pairs v0..3 -> K0..7,
// v4..7 -> K16..23; lanes 16-31 row=lane-16, v0..3 -> K8..15, v4..7 -> K24..31.
// Pairs (K, K+1) are contiguous -> single aligned 32-bit LDS load per pair.
__device__ __forceinline__ v16bf load_frag(const __bf16* base, int ld, int row0,
                                           int lane, int ks) {
  const __bf16* p = base + (size_t)(row0 + (lane & 15)) * ld + ks;
  const int kb = (lane < 16) ? 0 : 8;
  union { v16bf v; unsigned u[8]; } f;
#pragma unroll
  for (int e = 0; e < 8; e++) {
    int k = kb + ((e < 4) ? 2 * e : 2 * e + 8);
    f.u[e] = *reinterpret_cast<const unsigned*>(p + k);
  }
  return f.v;
}

// ---------------- bf16 WMMA GEMM:  C = A(MxK) @ BT(NxK)^T (+bias) -----------
// 256 threads = 8 waves; block tile 128x128, BK=64; wave tile 32x64 (2x4 WMMA
// per half-K => 16 v_wmma per K-tile).  Both operands staged to LDS with
// CDNA5 async global->LDS b128 copies (contiguous bf16), ASYNCcnt-tracked.
#define LDA_PAD 80   // 160B rows: b128-aligned, breaks bank-conflict stride
__global__ void __launch_bounds__(256)
gemm_wmma_bf16(const __bf16* __restrict__ A, const __bf16* __restrict__ BT,
               const float* __restrict__ bias, void* __restrict__ Cout,
               int M, int N, int K, int out_bf16) {
  __shared__ __bf16 As[128][LDA_PAD];
  __shared__ __bf16 Bs[128][LDA_PAD];

  const int tid  = threadIdx.x;
  const int lane = tid & 31;
  const int wave = tid >> 5;
  const int wm = (wave >> 1) << 5;   // 0,32,64,96
  const int wn = (wave & 1)  << 6;   // 0,64
  const int bM = blockIdx.y << 7;
  const int bN = blockIdx.x << 7;

  v8f acc[2][4] = {};

  // staging split: 1024 16B-vectors per operand tile, 4 per thread
  const int sr = tid >> 1;                 // 0..127 : tile row
  const int sk = (tid & 1) << 5;           // 0 / 32 : K sub-block (4 x 8 bf16)

  for (int k0 = 0; k0 < K; k0 += 64) {
    // ---- A tile: rows bM+sr, K k0+sk .. +31  (always in range) ----
#pragma unroll
    for (int i = 0; i < 4; i++) {
      unsigned long long ga =
          (unsigned long long)(const void*)(A + (size_t)(bM + sr) * K + k0 + sk + i * 8);
      unsigned la = (unsigned)(unsigned long long)(uintptr_t)&As[sr][sk + i * 8];
      asm volatile("global_load_async_to_lds_b128 %0, %1, off"
                   :: "v"(la), "v"(ga) : "memory");
    }
    // ---- B tile: rows bN+sr of BT (guard: logical column < N) ----
    if (bN + sr < N) {
#pragma unroll
      for (int i = 0; i < 4; i++) {
        unsigned long long ga =
            (unsigned long long)(const void*)(BT + (size_t)(bN + sr) * K + k0 + sk + i * 8);
        unsigned la = (unsigned)(unsigned long long)(uintptr_t)&Bs[sr][sk + i * 8];
        asm volatile("global_load_async_to_lds_b128 %0, %1, off"
                     :: "v"(la), "v"(ga) : "memory");
      }
    } else {
#pragma unroll
      for (int i = 0; i < 4; i++)
        *reinterpret_cast<uint4*>(&Bs[sr][sk + i * 8]) = make_uint4(0u, 0u, 0u, 0u);
    }
    if (k0 + 64 < K)   // warm next A block in GL2
      __builtin_prefetch(A + (size_t)(bM + sr) * K + k0 + 64 + sk, 0, 1);

    asm volatile("s_wait_asynccnt 0x0" ::: "memory");
    __syncthreads();

#pragma unroll
    for (int ks = 0; ks < 64; ks += 32) {
      v16bf af[2], bfr[4];
#pragma unroll
      for (int tm = 0; tm < 2; tm++)
        af[tm] = load_frag(&As[0][0], LDA_PAD, wm + (tm << 4), lane, ks);
#pragma unroll
      for (int tn = 0; tn < 4; tn++)
        bfr[tn] = load_frag(&Bs[0][0], LDA_PAD, wn + (tn << 4), lane, ks);
#pragma unroll
      for (int tm = 0; tm < 2; tm++)
#pragma unroll
        for (int tn = 0; tn < 4; tn++)
          acc[tm][tn] = __builtin_amdgcn_wmma_f32_16x16x32_bf16(
              false, af[tm], false, bfr[tn], (short)0, acc[tm][tn], false, false);
    }
    __syncthreads();
  }

  // ---- epilogue: C layout lane -> col lane&15, rows r + 8*(lane>>4) ----
#pragma unroll
  for (int tm = 0; tm < 2; tm++) {
#pragma unroll
    for (int tn = 0; tn < 4; tn++) {
      int gn = bN + wn + (tn << 4) + (lane & 15);
      if (gn < N) {
        float bv = bias ? bias[gn] : 0.0f;
        int gm0 = bM + wm + (tm << 4) + ((lane >> 4) << 3);
        if (out_bf16) {
          __bf16* C = (__bf16*)Cout;
#pragma unroll
          for (int r = 0; r < 8; r++)
            C[(size_t)(gm0 + r) * N + gn] = f2bf(acc[tm][tn][r] + bv);
        } else {
          float* C = (float*)Cout;
#pragma unroll
          for (int r = 0; r < 8; r++)
            C[(size_t)(gm0 + r) * N + gn] = acc[tm][tn][r] + bv;
        }
      }
    }
  }
}

// ---------------- f32 -> bf16 (8 elems/thread, b128 in / b128 out) ----------
__global__ void convert_bf16_kernel(const float* __restrict__ in,
                                    __bf16* __restrict__ out, int n8) {
  int j = blockIdx.x * blockDim.x + threadIdx.x;
  if (j >= n8) return;
  const float4* p = reinterpret_cast<const float4*>(in) + (size_t)j * 2;
  float4 a = p[0], b = p[1];
  union { unsigned short s[8]; uint4 u; } r;
  r.s[0] = bfbits(a.x); r.s[1] = bfbits(a.y); r.s[2] = bfbits(a.z); r.s[3] = bfbits(a.w);
  r.s[4] = bfbits(b.x); r.s[5] = bfbits(b.y); r.s[6] = bfbits(b.z); r.s[7] = bfbits(b.w);
  reinterpret_cast<uint4*>(out)[j] = r.u;
}

// ---------------- transpose + convert: in (RxC) f32 -> out (CxR) bf16 -------
__global__ void transpose_convert_kernel(const float* __restrict__ in,
                                         __bf16* __restrict__ out, int R, int C) {
  size_t j = (size_t)blockIdx.x * blockDim.x + threadIdx.x;
  if (j >= (size_t)R * C) return;
  int c = (int)(j / R), r = (int)(j % R);
  out[j] = f2bf(in[(size_t)r * C + c]);
}

// ---------------- batchnorm over batch axis ----------------
__global__ void bn_stats_kernel(const float* __restrict__ h,
                                float* __restrict__ mu, float* __restrict__ rstd) {
  int c = blockIdx.x * blockDim.x + threadIdx.x;
  if (c >= DIM_FEAT) return;
  float s = 0.f, s2 = 0.f;
  for (int r = 0; r < N_SAMPLES; r++) {
    float v = h[(size_t)r * DIM_FEAT + c];
    s += v; s2 += v * v;
  }
  float m   = s * (1.0f / N_SAMPLES);
  float var = s2 * (1.0f / N_SAMPLES) - m * m;
  mu[c] = m;
  rstd[c] = rsqrtf(var + 1e-5f);
}

// normalize + relu, emit bf16 activation for the next GEMM
__global__ void bn_apply_relu_kernel(const float* __restrict__ h,
                                     const float* __restrict__ mu,
                                     const float* __restrict__ rstd,
                                     const float* __restrict__ gamma,
                                     const float* __restrict__ beta,
                                     __bf16* __restrict__ h_bf) {
  size_t idx = (size_t)blockIdx.x * blockDim.x + threadIdx.x;
  int c = (int)(idx & (DIM_FEAT - 1));
  float v = (h[idx] - mu[c]) * rstd[c] * gamma[c] + beta[c];
  h_bf[idx] = f2bf(v > 0.f ? v : 0.f);
}

// ---------------- per-row L2 normalize, emit f32 (queue) + bf16 (GEMMs) -----
__global__ void row_l2_normalize_kernel(float* __restrict__ feat,
                                        __bf16* __restrict__ feat_bf) {
  __shared__ float red[DIM_EMB];
  int i = blockIdx.x, d = threadIdx.x;
  float v = feat[(size_t)i * DIM_EMB + d];
  red[d] = v * v;
  __syncthreads();
  for (int s = DIM_EMB / 2; s > 0; s >>= 1) {
    if (d < s) red[d] += red[d + s];
    __syncthreads();
  }
  float nv = v * rsqrtf(red[0]);
  feat[(size_t)i * DIM_EMB + d]    = nv;
  feat_bf[(size_t)i * DIM_EMB + d] = f2bf(nv);
}

// ---------------- exact top-k per row via bitonic sort in LDS ----------------
__global__ void __launch_bounds__(256)
topk_kernel(const float* __restrict__ sim_q, const int* __restrict__ labels,
            float* __restrict__ out_sim) {
  __shared__ float sv[8192];
  int i = blockIdx.x;
  int c = labels[i];
  const float* row = sim_q + (size_t)i * QC;

  for (int j = threadIdx.x; j < 8192; j += 256) {
    float v = -3.4e38f;
    if (j < NEG_LEN) {
      int col = j + (((j >> 3) >= c) ? QSZ : 0);   // neg_index[c][j]
      v = row[col];
    }
    sv[j] = v;
  }
  __syncthreads();
  for (int ksz = 2; ksz <= 8192; ksz <<= 1) {        // descending bitonic
    for (int jj = ksz >> 1; jj > 0; jj >>= 1) {
      for (int t = threadIdx.x; t < 8192; t += 256) {
        int ixj = t ^ jj;
        if (ixj > t) {
          float a = sv[t], b = sv[ixj];
          bool up = ((t & ksz) == 0);
          if (up ? (a < b) : (a > b)) { sv[t] = b; sv[ixj] = a; }
        }
      }
      __syncthreads();
    }
  }
  float* orow = out_sim + (size_t)i * LD_CON;
  for (int j = threadIdx.x; j < KNEG; j += 256)
    orow[2047 + KPOS + j] = sv[j];

  if (threadIdx.x == 0) {   // pos: 4 smallest of 8, ascending
    float p[QSZ];
    for (int q = 0; q < QSZ; q++) p[q] = row[c * QSZ + q];
    for (int a = 1; a < QSZ; a++) {
      float key = p[a]; int b = a - 1;
      while (b >= 0 && p[b] > key) { p[b + 1] = p[b]; b--; }
      p[b + 1] = key;
    }
    for (int q = 0; q < KPOS; q++) orow[2047 + q] = p[q];
  }
}

// ---------------- off-diagonal gather of sim_b + labels_con ----------------
__global__ void offdiag_labels_kernel(const float* __restrict__ sim_b,
                                      const int* __restrict__ labels,
                                      float* __restrict__ out_sim,
                                      float* __restrict__ out_lab) {
  int i = blockIdx.x;
  int li = labels[i];
  const float* srow = sim_b + (size_t)i * N_SAMPLES;
  float* so = out_sim + (size_t)i * LD_CON;
  float* lo = out_lab + (size_t)i * LD_CON;
  for (int j = threadIdx.x; j < LD_CON; j += 256) {
    if (j < N_SAMPLES - 1) {
      int col = j + (j >= i ? 1 : 0);
      so[j] = srow[col];
      lo[j] = (labels[col] == li) ? 1.0f : 0.0f;
    } else if (j < N_SAMPLES - 1 + KPOS) {
      lo[j] = 1.0f;
    } else {
      lo[j] = 0.0f;
    }
  }
}

// ---------------- queue ring-buffer update ----------------
__global__ void queue_copy_kernel(const float* __restrict__ src, float* __restrict__ dst) {
  size_t idx = (size_t)blockIdx.x * blockDim.x + threadIdx.x;
  dst[idx] = src[idx];
}

__global__ void __launch_bounds__(128)
queue_update_kernel(const int* __restrict__ labels, const int* __restrict__ queue_ptr,
                    const float* __restrict__ feat, float* __restrict__ out_queue,
                    float* __restrict__ out_ptr) {
  int c = blockIdx.x;
  int d = threadIdx.x;
  int ptr = queue_ptr[c];
  for (int i = 0; i < N_SAMPLES / 2; i++) {   // lax.scan order within class
    if (labels[i] == c) {
      int real = c * QSZ + ptr;
      out_queue[(size_t)d * QC + real] = feat[(size_t)i * DIM_EMB + d];
      ptr = (ptr + 1) & (QSZ - 1);
    }
  }
  if (d == 0) out_ptr[c] = (float)ptr;
}

// ---------------- launch ----------------
static inline char* bump(char*& p, size_t bytes) {
  char* r = p;
  p += (bytes + 255) & ~(size_t)255;
  return r;
}

extern "C" void kernel_launch(void* const* d_in, const int* in_sizes, int n_in,
                              void* d_out, int out_size, void* d_ws, size_t ws_size,
                              hipStream_t stream) {
  (void)in_sizes; (void)n_in; (void)out_size; (void)ws_size;
  const float* img        = (const float*)d_in[0];
  const int*   labels     = (const int*)  d_in[1];
  const float* queue_list = (const float*)d_in[2];
  const int*   queue_ptr  = (const int*)  d_in[3];
  // d_in[4] pos_index, d_in[5] neg_index derived analytically
  const float* enc_W  = (const float*)d_in[6];
  const float* enc_b  = (const float*)d_in[7];
  const float* mlp_W1 = (const float*)d_in[8];
  const float* mlp_b1 = (const float*)d_in[9];
  const float* gamma  = (const float*)d_in[10];
  const float* beta   = (const float*)d_in[11];
  const float* mlp_W2 = (const float*)d_in[12];
  const float* mlp_b2 = (const float*)d_in[13];
  const float* lin_W  = (const float*)d_in[14];
  const float* lin_b  = (const float*)d_in[15];

  // ---- workspace carve ----
  char* wp = (char*)d_ws;
  float*  h       = (float*) bump(wp, sizeof(float) * N_SAMPLES * DIM_FEAT);
  float*  feat    = (float*) bump(wp, sizeof(float) * N_SAMPLES * DIM_EMB);
  float*  mu      = (float*) bump(wp, sizeof(float) * DIM_FEAT);
  float*  rstd    = (float*) bump(wp, sizeof(float) * DIM_FEAT);
  float*  simq    = (float*) bump(wp, sizeof(float) * (size_t)N_SAMPLES * QC);
  float*  simb    = (float*) bump(wp, sizeof(float) * (size_t)N_SAMPLES * N_SAMPLES);
  __bf16* img_bf  = (__bf16*)bump(wp, 2ull * N_SAMPLES * DIM_IN);
  __bf16* mid_bf  = (__bf16*)bump(wp, 2ull * N_SAMPLES * DIM_FEAT);
  __bf16* h_bf    = (__bf16*)bump(wp, 2ull * N_SAMPLES * DIM_FEAT);
  __bf16* feat_bf = (__bf16*)bump(wp, 2ull * N_SAMPLES * DIM_EMB);
  __bf16* encWT   = (__bf16*)bump(wp, 2ull * DIM_FEAT * DIM_IN);     // 2048x1024
  __bf16* w1T     = (__bf16*)bump(wp, 2ull * DIM_FEAT * DIM_FEAT);   // 2048x2048
  __bf16* w2T     = (__bf16*)bump(wp, 2ull * DIM_EMB * DIM_FEAT);    // 128x2048
  __bf16* linWT   = (__bf16*)bump(wp, 2ull * NCLS * DIM_FEAT);       // 1000x2048
  __bf16* queueT  = (__bf16*)bump(wp, 2ull * QC * DIM_EMB);          // 8000x128

  // ---- output carve (tuple order) ----
  float* out       = (float*)d_out;
  float* out_sim   = out;
  float* out_lab   = out_sim   + (size_t)N_SAMPLES * LD_CON;
  float* out_logit = out_lab   + (size_t)N_SAMPLES * LD_CON;
  float* out_queue = out_logit + (size_t)N_SAMPLES * NCLS;
  float* out_ptr   = out_queue + (size_t)DIM_EMB * QC;

  // ---- operand precision/layout passes ----
  convert_bf16_kernel<<<dim3((N_SAMPLES * DIM_IN / 8 + 255) / 256), dim3(256), 0, stream>>>(
      img, img_bf, N_SAMPLES * DIM_IN / 8);
  transpose_convert_kernel<<<dim3((DIM_IN * DIM_FEAT + 255) / 256), dim3(256), 0, stream>>>(
      enc_W, encWT, DIM_IN, DIM_FEAT);
  transpose_convert_kernel<<<dim3((DIM_FEAT * DIM_FEAT + 255) / 256), dim3(256), 0, stream>>>(
      mlp_W1, w1T, DIM_FEAT, DIM_FEAT);
  transpose_convert_kernel<<<dim3((DIM_FEAT * DIM_EMB + 255) / 256), dim3(256), 0, stream>>>(
      mlp_W2, w2T, DIM_FEAT, DIM_EMB);
  transpose_convert_kernel<<<dim3((DIM_FEAT * NCLS + 255) / 256), dim3(256), 0, stream>>>(
      lin_W, linWT, DIM_FEAT, NCLS);
  transpose_convert_kernel<<<dim3((DIM_EMB * QC + 255) / 256), dim3(256), 0, stream>>>(
      queue_list, queueT, DIM_EMB, QC);

  const dim3 blk(256);
  const int gy = N_SAMPLES / 128;
  // 1) mid = img @ enc_W + enc_b      (emit bf16 directly)
  gemm_wmma_bf16<<<dim3(DIM_FEAT / 128, gy), blk, 0, stream>>>(
      img_bf, encWT, enc_b, mid_bf, N_SAMPLES, DIM_FEAT, DIM_IN, 1);
  // 2) h = mid @ mlp_W1 + mlp_b1      (f32 for BN stats)
  gemm_wmma_bf16<<<dim3(DIM_FEAT / 128, gy), blk, 0, stream>>>(
      mid_bf, w1T, mlp_b1, h, N_SAMPLES, DIM_FEAT, DIM_FEAT, 0);
  // 3) logit_cls = mid @ lin_W + lin_b
  gemm_wmma_bf16<<<dim3((NCLS + 127) / 128, gy), blk, 0, stream>>>(
      mid_bf, linWT, lin_b, out_logit, N_SAMPLES, NCLS, DIM_FEAT, 0);
  // 4-5) batchnorm + relu -> h_bf
  bn_stats_kernel<<<dim3(DIM_FEAT / 256), dim3(256), 0, stream>>>(h, mu, rstd);
  bn_apply_relu_kernel<<<dim3((N_SAMPLES * DIM_FEAT) / 256), dim3(256), 0, stream>>>(
      h, mu, rstd, gamma, beta, h_bf);
  // 6) feat = h @ mlp_W2 + mlp_b2
  gemm_wmma_bf16<<<dim3(DIM_EMB / 128, gy), blk, 0, stream>>>(
      h_bf, w2T, mlp_b2, feat, N_SAMPLES, DIM_EMB, DIM_FEAT, 0);
  // 7) L2 normalize rows -> feat (f32) + feat_bf
  row_l2_normalize_kernel<<<dim3(N_SAMPLES), dim3(DIM_EMB), 0, stream>>>(feat, feat_bf);
  // 8) sim_q = feat @ queue_list   (BT = queueT, 8000x128)
  gemm_wmma_bf16<<<dim3((QC + 127) / 128, gy), blk, 0, stream>>>(
      feat_bf, queueT, nullptr, simq, N_SAMPLES, QC, DIM_EMB, 0);
  // 9) sim_b = feat @ feat^T       (BT = feat itself)
  gemm_wmma_bf16<<<dim3(N_SAMPLES / 128, gy), blk, 0, stream>>>(
      feat_bf, feat_bf, nullptr, simb, N_SAMPLES, N_SAMPLES, DIM_EMB, 0);
  // 10) exact sorted top-k selection
  topk_kernel<<<dim3(N_SAMPLES), dim3(256), 0, stream>>>(simq, labels, out_sim);
  // 11) off-diagonal sims + labels_con
  offdiag_labels_kernel<<<dim3(N_SAMPLES), dim3(256), 0, stream>>>(
      simb, labels, out_sim, out_lab);
  // 12-13) queue ring-buffer update
  queue_copy_kernel<<<dim3((DIM_EMB * QC) / 256), dim3(256), 0, stream>>>(
      queue_list, out_queue);
  queue_update_kernel<<<dim3(NCLS), dim3(128), 0, stream>>>(
      labels, queue_ptr, feat, out_queue, out_ptr);
}
